// GraphAttentionLayer_42253888258789
// MI455X (gfx1250) — matrix operand order repaired and checked by reading, not compiled, not run on hardware
//
#include <hip/hip_runtime.h>

typedef __attribute__((ext_vector_type(16))) __bf16 v16bf;
typedef __attribute__((ext_vector_type(8)))  float  v8f;
typedef __attribute__((ext_vector_type(4)))  float  v4f;
typedef __attribute__((ext_vector_type(4)))  unsigned int v4u;
typedef __attribute__((ext_vector_type(2)))  unsigned int v2u;

#define N_NODES 8192
#define F_IN    256
#define F_OUTD  64
#define NEG_SLOPE 0.2f

static __device__ __forceinline__ unsigned short f32_to_bf16(float f) {
    union { float f; unsigned int u; } v; v.f = f;
    unsigned int r = v.u + 0x7FFFu + ((v.u >> 16) & 1u);  // round-to-nearest-even
    return (unsigned short)(r >> 16);
}

// 16 bf16 per lane = one WMMA A/B operand (8 VGPRs), with raw views for packing.
union ATile { v16bf v; unsigned short u[16]; v4u q[2]; };

// ---------------- kernel 1: h (f32) -> h bf16 ----------------
__global__ void cvt_h_bf16(const float* __restrict__ h,
                           unsigned short* __restrict__ hb, int n4) {
    for (int i = blockIdx.x * blockDim.x + threadIdx.x; i < n4;
         i += gridDim.x * blockDim.x) {
        v4f x = reinterpret_cast<const v4f*>(h)[i];
        v2u o;
        o.x = (unsigned)f32_to_bf16(x.x) | ((unsigned)f32_to_bf16(x.y) << 16);
        o.y = (unsigned)f32_to_bf16(x.z) | ((unsigned)f32_to_bf16(x.w) << 16);
        reinterpret_cast<v2u*>(hb)[i] = o;
    }
}

// ---------------- kernel 2: W [256][64] -> WT bf16 [64][256] ----------------
__global__ void cvt_w_t(const float* __restrict__ W, unsigned short* __restrict__ WT) {
    int idx = blockIdx.x * blockDim.x + threadIdx.x;
    if (idx < F_OUTD * F_IN) {
        int f = idx >> 8, k = idx & 255;
        WT[f * F_IN + k] = f32_to_bf16(W[k * F_OUTD + f]);
    }
}

// ---------------- kernel 3: Wh = h@W (WMMA bf16), fused s1/s2 + WhT bf16 ----------------
__global__ void __launch_bounds__(128) gemm_wh(
    const unsigned short* __restrict__ hb,   // [8192][256] bf16
    const unsigned short* __restrict__ WT,   // [64][256]  bf16
    const float* __restrict__ a,             // [128] f32
    unsigned short* __restrict__ WhT,        // [64][8192] bf16 out
    float* __restrict__ s1, float* __restrict__ s2)
{
    const int lane = threadIdx.x & 31;
    const int wid  = (blockIdx.x * blockDim.x + threadIdx.x) >> 5;
    const int m0   = wid * 16;
    const int lh   = lane >> 4;      // half of wave (K-split per ISA layout)
    const int ln   = lane & 15;

    v8f c[4] = {};                   // 16x16 f32 accum for n-tiles 0..3
    const int r = m0 + ln;           // A: lanes 0-15 -> M=lane, 16-31 -> M=lane-16
    for (int kc = 0; kc < F_IN; kc += 32) {
        ATile A;
        const int ka = kc + (lh << 3);     // K runs: [ka..ka+7], [ka+16..ka+23]
        A.q[0] = *reinterpret_cast<const v4u*>(hb + r * F_IN + ka);
        A.q[1] = *reinterpret_cast<const v4u*>(hb + r * F_IN + ka + 16);
        const int kb = kc + (lh << 3);     // B: K runs [kb..kb+7], [kb+16..kb+23] (same as A)
#pragma unroll
        for (int t = 0; t < 4; ++t) {
            ATile B;
            const unsigned short* bp = WT + (t * 16 + ln) * F_IN + kb;
            B.q[0] = reinterpret_cast<const v4u*>(bp)[0];
            B.q[1] = reinterpret_cast<const v4u*>(bp)[2];
            c[t] = __builtin_amdgcn_wmma_f32_16x16x32_bf16(
                       false, A.v, false, B.v, (short)0, c[t], false, false);
        }
    }

    // epilogue: write WhT bf16 (contiguous 8 rows per lane) + s1/s2 projections
    float a1v[4], a2v[4];
#pragma unroll
    for (int t = 0; t < 4; ++t) {
        a1v[t] = a[t * 16 + ln];
        a2v[t] = a[64 + t * 16 + ln];
    }
    float p1[8] = {}, p2[8] = {};
#pragma unroll
    for (int t = 0; t < 4; ++t) {
        v4u pk;
        pk.x = (unsigned)f32_to_bf16(c[t][0]) | ((unsigned)f32_to_bf16(c[t][1]) << 16);
        pk.y = (unsigned)f32_to_bf16(c[t][2]) | ((unsigned)f32_to_bf16(c[t][3]) << 16);
        pk.z = (unsigned)f32_to_bf16(c[t][4]) | ((unsigned)f32_to_bf16(c[t][5]) << 16);
        pk.w = (unsigned)f32_to_bf16(c[t][6]) | ((unsigned)f32_to_bf16(c[t][7]) << 16);
        *reinterpret_cast<v4u*>(WhT + (t * 16 + ln) * N_NODES + m0 + (lh << 3)) = pk;
#pragma unroll
        for (int vv = 0; vv < 8; ++vv) {
            p1[vv] += c[t][vv] * a1v[t];
            p2[vv] += c[t][vv] * a2v[t];
        }
    }
    // reduce across the 16 lanes of each half (row data lives within a half)
#pragma unroll
    for (int mask = 1; mask <= 8; mask <<= 1) {
#pragma unroll
        for (int vv = 0; vv < 8; ++vv) {
            p1[vv] += __shfl_xor(p1[vv], mask, 32);
            p2[vv] += __shfl_xor(p2[vv], mask, 32);
        }
    }
    if (ln == 0) {
        const int rb = m0 + (lh << 3);
#pragma unroll
        for (int vv = 0; vv < 8; ++vv) { s1[rb + vv] = p1[vv]; s2[rb + vv] = p2[vv]; }
    }
}

// ---------------- kernel 4: scalar max over s2 ----------------
__global__ void reduce_max(const float* __restrict__ s2, float* __restrict__ s2max) {
    __shared__ float red[1024];
    float m = -3.402823466e38f;
    for (int i = threadIdx.x; i < N_NODES; i += 1024) m = fmaxf(m, s2[i]);
    red[threadIdx.x] = m;
    __syncthreads();
    for (int s = 512; s > 0; s >>= 1) {
        if (threadIdx.x < s) red[threadIdx.x] = fmaxf(red[threadIdx.x], red[threadIdx.x + s]);
        __syncthreads();
    }
    if (threadIdx.x == 0) *s2max = red[0];
}

// ---------------- kernel 5: fused softmax(e) @ Wh, single pass ----------------
// 64-j pipeline stages: Whᵀ[.., j0..j0+63] (8 KB) staged block-wide into LDS with
// async copies (double-buffered). All 4 waves consume the same B tiles via
// ds_load_b128; exp/P-tile VALU work overlaps the in-flight DMA.
__global__ void __launch_bounds__(128) attn(
    const unsigned short* __restrict__ WhT,  // [64][8192] bf16
    const float* __restrict__ s1, const float* __restrict__ s2,
    const float* __restrict__ s2maxp, float* __restrict__ out)
{
    __shared__ __align__(16) unsigned short smemB[2 * 4096];  // 2 buffers x 8 KB

    const int tid  = threadIdx.x;
    const int lane = tid & 31;
    const int wid  = (blockIdx.x * blockDim.x + tid) >> 5;
    const int m0   = wid * 16;
    const int lh   = lane >> 4;
    const int ln   = lane & 15;

    // ---- async-copy geometry: 8 KB stage = 128 threads x 4 units x 16 B ----
    // stage layout [n][k]: flat byte f -> row n = f/128, byte-in-row = f%128
    const unsigned ldsBase = (unsigned)(size_t)(&smemB[0]);   // low 32 bits = LDS offset
    int fb[4], vc[4];
#pragma unroll
    for (int u = 0; u < 4; ++u) {
        fb[u] = u * 2048 + tid * 16;
        vc[u] = (fb[u] >> 7) * (N_NODES * 2) + (fb[u] & 127); // global byte offset at j0=0
    }
    const unsigned long long sbase = (unsigned long long)(size_t)WhT;

    auto issue_stage = [&](int j0, int buf) {
#pragma unroll
        for (int u = 0; u < 4; ++u) {
            unsigned l = ldsBase + buf * 8192 + fb[u];
            int v = vc[u] + j0 * 2;
            asm volatile("global_load_async_to_lds_b128 %0, %1, %2"
                         :: "v"(l), "v"(v), "s"(sbase) : "memory");
        }
    };

    const float s2max = *s2maxp;
    const int r = m0 + ln;
    const float s1r = s1[r];
    const float tm  = s1r + s2max;
    const float mr  = fmaxf(tm, NEG_SLOPE * tm);   // lrelu monotone -> exact row max
    const float LOG2E = 1.44269504088896f;

    v8f c[4] = {};
    float lsum = 0.f;

    issue_stage(0, 0);                 // prologue prefetch
    int buf = 0;
    for (int j0 = 0; j0 < N_NODES; j0 += 64) {
        // ---- build two P tiles (j0, j0+32) in registers; overlaps in-flight DMA
        ATile P[2];
#pragma unroll
        for (int s = 0; s < 2; ++s) {
            const int klo = j0 + s * 32 + (lh << 3);
            v4f se0 = *reinterpret_cast<const v4f*>(s2 + klo);
            v4f se1 = *reinterpret_cast<const v4f*>(s2 + klo + 4);
            v4f se2 = *reinterpret_cast<const v4f*>(s2 + klo + 16);
            v4f se3 = *reinterpret_cast<const v4f*>(s2 + klo + 20);
            float se[16] = { se0.x, se0.y, se0.z, se0.w,  se1.x, se1.y, se1.z, se1.w,
                             se2.x, se2.y, se2.z, se2.w,  se3.x, se3.y, se3.z, se3.w };
#pragma unroll
            for (int i = 0; i < 16; ++i) {
                float e = s1r + se[i];
                e = fmaxf(e, NEG_SLOPE * e);                        // LeakyReLU(0.2)
                float p = __builtin_amdgcn_exp2f(LOG2E * (e - mr)); // v_exp_f32 (TRANS)
                lsum += p;
                P[s].u[i] = f32_to_bf16(p);
            }
        }

        // ---- pipeline control (one barrier pair per 64 j)
        __syncthreads();               // all waves finished reading buf^1 -> refill it
        if (j0 + 64 < N_NODES) {
            issue_stage(j0 + 64, buf ^ 1);
            asm volatile("s_wait_asynccnt 0x4" ::: "memory");  // oldest 4 (= cur stage) done
        } else {
            asm volatile("s_wait_asynccnt 0x0" ::: "memory");
        }
        __syncthreads();               // every wave's view of cur stage is complete

        // ---- 8 WMMAs: two P[16x32] @ Wh[32x64] sub-steps from LDS
        const int bufo = buf * 4096;   // ushort units
#pragma unroll
        for (int s = 0; s < 2; ++s) {
#pragma unroll
            for (int t = 0; t < 4; ++t) {
                ATile B;
                const int bo = bufo + (t * 16 + ln) * 64 + s * 32 + (lh << 3);
                B.q[0] = *reinterpret_cast<const v4u*>(&smemB[bo]);
                B.q[1] = *reinterpret_cast<const v4u*>(&smemB[bo + 16]);
                c[t] = __builtin_amdgcn_wmma_f32_16x16x32_bf16(
                           false, P[s].v, false, B.v, (short)0, c[t], false, false);
            }
        }
        buf ^= 1;
    }

    // row sums: each half covered disjoint K ranges of the same row (ln)
    float tot = lsum + __shfl_xor(lsum, 16, 32);
    float linv[8];
#pragma unroll
    for (int vv = 0; vv < 8; ++vv) {
        float lv = __shfl(tot, vv + (lh << 3), 32);
        linv[vv] = __builtin_amdgcn_rcpf(lv);
    }
#pragma unroll
    for (int t = 0; t < 4; ++t)
#pragma unroll
        for (int vv = 0; vv < 8; ++vv)
            out[(m0 + vv + (lh << 3)) * F_OUTD + t * 16 + ln] = c[t][vv] * linv[vv];
}

// ---------------- host-side launch ----------------
extern "C" void kernel_launch(void* const* d_in, const int* in_sizes, int n_in,
                              void* d_out, int out_size, void* d_ws, size_t ws_size,
                              hipStream_t stream) {
    (void)in_sizes; (void)n_in; (void)out_size; (void)ws_size;
    const float* h = (const float*)d_in[0];
    // d_in[1] = adj (bool) — unused by the reference forward
    const float* W = (const float*)d_in[2];
    const float* a = (const float*)d_in[3];
    float* out = (float*)d_out;

    char* ws = (char*)d_ws;
    unsigned short* hb  = (unsigned short*)(ws);                                  // 4 MiB
    unsigned short* WT  = (unsigned short*)(ws + (4u << 20));                     // 32 KiB
    unsigned short* WhT = (unsigned short*)(ws + (4u << 20) + (32u << 10));       // 1 MiB
    float* s1  = (float*)(ws + (4u << 20) + (32u << 10) + (1u << 20));            // 32 KiB
    float* s2  = s1 + N_NODES;                                                    // 32 KiB
    float* s2m = s2 + N_NODES;                                                    // 4 B

    cvt_h_bf16<<<2048, 256, 0, stream>>>(h, hb, (N_NODES * F_IN) / 4);
    cvt_w_t  <<<64, 256, 0, stream>>>(W, WT);
    gemm_wh  <<<128, 128, 0, stream>>>(hb, WT, a, WhT, s1, s2);   // 512 waves, 16 rows each
    reduce_max<<<1, 1024, 0, stream>>>(s2, s2m);
    attn     <<<128, 128, 0, stream>>>(WhT, s1, s2, s2m, out);    // 512 waves, 16 rows each
}
